// BasicMoe_30468497998332
// MI455X (gfx1250) — matrix operand order repaired and checked by reading, compiled, not run on hardware
//
#include <hip/hip_runtime.h>
#include <hip/hip_bf16.h>

// MoE top-2 gathered forward for MI455X (gfx1250), bf16 WMMA path.
// d_out layout: out[4096*1024] f32, then router_logits[4096*8] f32.
// d_ws usage (~42.2 MB): counts(32B) | offsets(32B) | lists(128KB) |
//   wlists(128KB) | xb bf16 (8MB) | hb bf16 (32MB)

#define T_TOK 4096
#define D_DIM 1024
#define F_DIM 2048
#define E_EXP 8

typedef __attribute__((ext_vector_type(16))) __bf16 v16bf;
typedef __attribute__((ext_vector_type(8)))  float  v8f;

union Frag16 { uint4 q[2]; v16bf v; };

__device__ __forceinline__ unsigned short f2bf(float f) {
  unsigned u = __float_as_uint(f);
  u += 0x7FFFu + ((u >> 16) & 1u);          // round-to-nearest-even
  return (unsigned short)(u >> 16);
}

// Async global -> LDS direct copy (CDNA5 GLOBAL_LOAD_ASYNC_TO_LDS_B128,
// tracked by ASYNCcnt). ldsOff = workgroup-relative LDS byte offset
// (= low 32 bits of a generic pointer into __shared__, per the LDS aperture).
__device__ __forceinline__ void async_load_lds_b128(unsigned ldsOff, const void* g) {
  asm volatile("global_load_async_to_lds_b128 %0, %1, off"
               :: "v"(ldsOff), "v"(g) : "memory");
}
__device__ __forceinline__ void wait_async_zero() {
  asm volatile("s_wait_asynccnt 0" ::: "memory");
}
__device__ __forceinline__ unsigned lds_off(const void* p) {
  return (unsigned)(unsigned long long)p;
}

// ---------------- init: zero output accumulation region + expert counters ---
__global__ void zero_kernel(float* __restrict__ out, int n, int* __restrict__ counts) {
  int i = blockIdx.x * blockDim.x + threadIdx.x;
  if (i < E_EXP) counts[i] = 0;
  for (int k = i; k < n; k += gridDim.x * blockDim.x) out[k] = 0.f;
}

// ---------------- convert hidden states to bf16 ----------------------------
__global__ void cvt_kernel(const float* __restrict__ x, unsigned short* __restrict__ xb, int n) {
  int i = blockIdx.x * blockDim.x + threadIdx.x;
  for (int k = i; k < n; k += gridDim.x * blockDim.x) xb[k] = f2bf(x[k]);
}

// ---------------- router: logits, softmax, top-2, expert lists --------------
__global__ void router_kernel(const float* __restrict__ x, const float* __restrict__ gw,
                              float* __restrict__ rlog, int* __restrict__ lists,
                              float* __restrict__ wlists, int* __restrict__ counts) {
  const int t    = blockIdx.x;
  const int tid  = threadIdx.x;
  const int w    = tid >> 5;
  const int lane = tid & 31;
  __shared__ float sl[E_EXP];

  const float* xr = x  + (size_t)t * D_DIM;
  const float* gr = gw + (size_t)w * D_DIM;
  float acc = 0.f;
  for (int d = lane; d < D_DIM; d += 32) acc += xr[d] * gr[d];
#pragma unroll
  for (int off = 16; off > 0; off >>= 1) acc += __shfl_xor(acc, off, 32);
  if (lane == 0) sl[w] = acc;
  __syncthreads();

  if (tid == 0) {
    float l[E_EXP], m = -1e30f;
#pragma unroll
    for (int e = 0; e < E_EXP; ++e) { l[e] = sl[e]; rlog[t * E_EXP + e] = l[e]; m = fmaxf(m, l[e]); }
    float p[E_EXP], s = 0.f;
#pragma unroll
    for (int e = 0; e < E_EXP; ++e) { p[e] = __expf(l[e] - m); s += p[e]; }
    float inv = 1.f / s;
#pragma unroll
    for (int e = 0; e < E_EXP; ++e) p[e] *= inv;
    // top-2, lowest index wins ties (jax top_k semantics)
    int i1 = 0;
#pragma unroll
    for (int e = 1; e < E_EXP; ++e) if (p[e] > p[i1]) i1 = e;
    int i2 = (i1 == 0) ? 1 : 0;
#pragma unroll
    for (int e = 0; e < E_EXP; ++e) if (e != i1 && p[e] > p[i2]) i2 = e;
    float denom = p[i1] + p[i2];
    float wa = p[i1] / denom, wb = p[i2] / denom;
    int pos1 = atomicAdd(&counts[i1], 1);
    lists[i1 * T_TOK + pos1] = t; wlists[i1 * T_TOK + pos1] = wa;
    int pos2 = atomicAdd(&counts[i2], 1);
    lists[i2 * T_TOK + pos2] = t; wlists[i2 * T_TOK + pos2] = wb;
  }
}

// ---------------- 8-entry exclusive prefix sum ------------------------------
__global__ void offsets_kernel(const int* __restrict__ counts, int* __restrict__ offsets) {
  if (threadIdx.x == 0) {
    int s = 0;
    for (int e = 0; e < E_EXP; ++e) { offsets[e] = s; s += counts[e]; }
  }
}

// ---------------- up-proj grouped GEMM + SiLU -------------------------------
// Per expert e: rows = its gathered tokens, A = xb[tok, D] bf16, B = w1[e] (D x F) f32
// Tile 128x128, K-step 32, 8 waves each computing 32x64 via v_wmma_f32_16x16x32_bf16.
// A tile staged via async global->LDS; B tile converted f32->bf16 into LDS transposed.
__global__ void __launch_bounds__(256)
moe_up_kernel(const unsigned short* __restrict__ xb, const float* __restrict__ w1,
              const int* __restrict__ lists, const int* __restrict__ counts,
              const int* __restrict__ offsets, unsigned short* __restrict__ hb) {
  const int e = blockIdx.z;
  const int cnt = counts[e];
  const int rowBase = blockIdx.y * 128;
  if (rowBase >= cnt) return;                       // block-uniform: EXEC stays all-1s
  const int colBase = blockIdx.x * 128;
  const int tid = threadIdx.x;

  __shared__ __align__(16) unsigned short As[128][40];  // [m][k] bf16, +8 pad
  __shared__ __align__(16) unsigned short Bs[128][40];  // [n][k] bf16 (transposed)
  __shared__ int rowTok[128];

  if (tid < 128) {
    int r = rowBase + tid;
    rowTok[tid] = lists[e * T_TOK + ((r < cnt) ? r : (cnt - 1))];
  }
  __syncthreads();

  const int wid = tid >> 5, lane = tid & 31;
  const int lm = lane & 15, half = lane >> 4;
  const int wm = (wid & 3) * 32;   // wave M offset in tile (4 waves along M)
  const int wn = (wid >> 2) * 64;  // wave N offset in tile (2 waves along N)

  v8f vzero = {};
  v8f acc[2][4];
#pragma unroll
  for (int i = 0; i < 2; ++i)
#pragma unroll
    for (int j = 0; j < 4; ++j) acc[i][j] = vzero;

  const float* w1e = w1 + (size_t)e * D_DIM * F_DIM;
  const int k2 = (tid >> 4) & 15;   // K pair id: k = 2*k2
  const int ng = (tid & 15) * 8;    // N group base (8 columns)

  for (int kk = 0; kk < D_DIM; kk += 32) {
    // A tile: gathered bf16 rows -> LDS, async direct copy (ASYNCcnt path)
#pragma unroll
    for (int it = 0; it < 2; ++it) {
      int idx = tid + it * 256;
      int r = idx >> 2, c = (idx & 3) * 8;
      async_load_lds_b128(lds_off(&As[r][c]),
                          xb + (size_t)rowTok[r] * D_DIM + kk + c);
    }
    // B tile: 2 K-rows x 8 N-cols per thread, pack bf16 pairs, ds_store_b32
    {
      const int k = 2 * k2;
      const float* s0 = w1e + (size_t)(kk + k) * F_DIM + colBase + ng;
      const float* s1 = s0 + F_DIM;
      if (kk + 32 < D_DIM) __builtin_prefetch(s0 + 32 * F_DIM, 0, 1);
#pragma unroll
      for (int j = 0; j < 8; ++j) {
        unsigned pk = (unsigned)f2bf(s0[j]) | ((unsigned)f2bf(s1[j]) << 16);
        *(unsigned*)&Bs[ng + j][k] = pk;
      }
    }
    wait_async_zero();
    __syncthreads();

    Frag16 a[2], b[4];
#pragma unroll
    for (int i = 0; i < 2; ++i) {
      int row = wm + i * 16 + lm;
      a[i].q[0] = *(const uint4*)&As[row][8 * half];        // K = 8*half .. +7  -> VGPR0-3
      a[i].q[1] = *(const uint4*)&As[row][16 + 8 * half];   // K = 16+8*half ..  -> VGPR4-7
    }
#pragma unroll
    for (int j = 0; j < 4; ++j) {
      int ncol = wn + j * 16 + lm;
      b[j].q[0] = *(const uint4*)&Bs[ncol][8 * half];       // K = 8*half .. +7
      b[j].q[1] = *(const uint4*)&Bs[ncol][16 + 8 * half];  // K = 16+8*half .. +7
    }
#pragma unroll
    for (int i = 0; i < 2; ++i)
#pragma unroll
      for (int j = 0; j < 4; ++j)
        acc[i][j] = __builtin_amdgcn_wmma_f32_16x16x32_bf16(
            false, a[i].v, false, b[j].v, (short)0, acc[i][j], false, false);
    __syncthreads();
  }

  // SiLU + bf16 store into compacted h buffer
  const int hbBase = offsets[e];
#pragma unroll
  for (int i = 0; i < 2; ++i)
#pragma unroll
    for (int j = 0; j < 4; ++j)
#pragma unroll
      for (int r = 0; r < 8; ++r) {
        int grow = rowBase + wm + i * 16 + r + 8 * half;
        if (grow < cnt) {
          float v = acc[i][j][r];
          float s = v / (1.f + __expf(-v));
          hb[(size_t)(hbBase + grow) * F_DIM + colBase + wn + j * 16 + lm] = f2bf(s);
        }
      }
}

// ---------------- down-proj grouped GEMM + weighted scatter-add -------------
__global__ void __launch_bounds__(256)
moe_down_kernel(const unsigned short* __restrict__ hb, const float* __restrict__ w2,
                const int* __restrict__ lists, const float* __restrict__ wlists,
                const int* __restrict__ counts, const int* __restrict__ offsets,
                float* __restrict__ out) {
  const int e = blockIdx.z;
  const int cnt = counts[e];
  const int rowBase = blockIdx.y * 128;
  if (rowBase >= cnt) return;
  const int colBase = blockIdx.x * 128;
  const int tid = threadIdx.x;

  __shared__ __align__(16) unsigned short As[128][40];
  __shared__ __align__(16) unsigned short Bs[128][40];
  __shared__ int   rowTok[128];
  __shared__ float rowW[128];

  if (tid < 128) {
    int r = rowBase + tid;
    int rc = (r < cnt) ? r : (cnt - 1);
    rowTok[tid] = lists[e * T_TOK + rc];
    rowW[tid]   = wlists[e * T_TOK + rc];
  }
  __syncthreads();

  const int wid = tid >> 5, lane = tid & 31;
  const int lm = lane & 15, half = lane >> 4;
  const int wm = (wid & 3) * 32;
  const int wn = (wid >> 2) * 64;

  v8f vzero = {};
  v8f acc[2][4];
#pragma unroll
  for (int i = 0; i < 2; ++i)
#pragma unroll
    for (int j = 0; j < 4; ++j) acc[i][j] = vzero;

  const float* w2e = w2 + (size_t)e * F_DIM * D_DIM;
  const int hbBase = offsets[e];
  const int k2 = (tid >> 4) & 15;
  const int ng = (tid & 15) * 8;

  for (int kk = 0; kk < F_DIM; kk += 32) {
    // A tile: contiguous compacted h rows, async direct copy to LDS
#pragma unroll
    for (int it = 0; it < 2; ++it) {
      int idx = tid + it * 256;
      int r = idx >> 2, c = (idx & 3) * 8;
      int rc = rowBase + r; if (rc >= cnt) rc = cnt - 1;   // stay inside this expert
      async_load_lds_b128(lds_off(&As[r][c]),
                          hb + (size_t)(hbBase + rc) * F_DIM + kk + c);
    }
    // B tile: w2[e] f32 -> bf16 packed pairs, transposed
    {
      const int k = 2 * k2;
      const float* s0 = w2e + (size_t)(kk + k) * D_DIM + colBase + ng;
      const float* s1 = s0 + D_DIM;
      if (kk + 32 < F_DIM) __builtin_prefetch(s0 + 32 * D_DIM, 0, 1);
#pragma unroll
      for (int j = 0; j < 8; ++j) {
        unsigned pk = (unsigned)f2bf(s0[j]) | ((unsigned)f2bf(s1[j]) << 16);
        *(unsigned*)&Bs[ng + j][k] = pk;
      }
    }
    wait_async_zero();
    __syncthreads();

    Frag16 a[2], b[4];
#pragma unroll
    for (int i = 0; i < 2; ++i) {
      int row = wm + i * 16 + lm;
      a[i].q[0] = *(const uint4*)&As[row][8 * half];
      a[i].q[1] = *(const uint4*)&As[row][16 + 8 * half];
    }
#pragma unroll
    for (int j = 0; j < 4; ++j) {
      int ncol = wn + j * 16 + lm;
      b[j].q[0] = *(const uint4*)&Bs[ncol][8 * half];
      b[j].q[1] = *(const uint4*)&Bs[ncol][16 + 8 * half];
    }
#pragma unroll
    for (int i = 0; i < 2; ++i)
#pragma unroll
      for (int j = 0; j < 4; ++j)
        acc[i][j] = __builtin_amdgcn_wmma_f32_16x16x32_bf16(
            false, a[i].v, false, b[j].v, (short)0, acc[i][j], false, false);
    __syncthreads();
  }

  // weight * result, scatter-add into out (two experts per token -> atomics)
#pragma unroll
  for (int i = 0; i < 2; ++i)
#pragma unroll
    for (int j = 0; j < 4; ++j)
#pragma unroll
      for (int r = 0; r < 8; ++r) {
        int lrow = wm + i * 16 + r + 8 * half;
        int grow = rowBase + lrow;
        if (grow < cnt) {
          float v = acc[i][j][r] * rowW[lrow];
          atomicAdd(&out[(size_t)rowTok[lrow] * D_DIM + colBase + wn + j * 16 + lm], v);
        }
      }
}

// ---------------------------------------------------------------------------
extern "C" void kernel_launch(void* const* d_in, const int* in_sizes, int n_in,
                              void* d_out, int out_size, void* d_ws, size_t ws_size,
                              hipStream_t stream) {
  (void)in_sizes; (void)n_in; (void)out_size; (void)ws_size;
  const float* x  = (const float*)d_in[0];   // [B,S,D] f32
  const float* gw = (const float*)d_in[1];   // [E,D]   f32
  const float* w1 = (const float*)d_in[2];   // [E,D,F] f32
  const float* w2 = (const float*)d_in[3];   // [E,F,D] f32

  float* out  = (float*)d_out;
  float* rlog = out + (size_t)T_TOK * D_DIM; // tuple output #2

  char* ws = (char*)d_ws;
  int*            counts  = (int*)ws;                         // 8 ints
  int*            offsets = (int*)(ws + 32);                  // 8 ints
  int*            lists   = (int*)(ws + 256);                 // [E][T]
  float*          wlists  = (float*)(ws + 256 + 131072);      // [E][T]
  unsigned short* xb      = (unsigned short*)(ws + 262400);   // [T][D] bf16
  unsigned short* hb      = (unsigned short*)(ws + 262400 + 8388608); // [T*2][F] bf16

  zero_kernel<<<4096, 256, 0, stream>>>(out, T_TOK * D_DIM, counts);
  cvt_kernel<<<4096, 256, 0, stream>>>(x, xb, T_TOK * D_DIM);
  router_kernel<<<T_TOK, 256, 0, stream>>>(x, gw, rlog, lists, wlists, counts);
  offsets_kernel<<<1, 32, 0, stream>>>(counts, offsets);
  moe_up_kernel<<<dim3(F_DIM / 128, T_TOK / 128, E_EXP), 256, 0, stream>>>(
      xb, w1, lists, counts, offsets, hb);
  moe_down_kernel<<<dim3(D_DIM / 128, T_TOK / 128, E_EXP), 256, 0, stream>>>(
      hb, w2, lists, wlists, counts, offsets, out);
}
